// DoubleMaskedChamferDistanceVisual_58308476010764
// MI455X (gfx1250) — matrix operand order repaired and checked by reading, not compiled, not run on hardware
//
#include <hip/hip_runtime.h>
#include <hip/hip_bf16.h>
#include <float.h>

// ---------------------------------------------------------------------------
// DoubleMaskedChamferDistance for MI455X (gfx1250, wave32, WMMA)
//   pairwise[b,n,m] = ||v_n||^2 - 2 v_n.l_m + ||l_m||^2
//   out[b] = sum_valid_rows(min over valid cols)/Nv + sum_valid_cols(min over valid rows)/Nl
// GEMM via v_wmma_f32_16x16x32_bf16 with bf16 hi/lo split (3 wmma) for ~fp32 accuracy.
// ---------------------------------------------------------------------------

typedef __attribute__((ext_vector_type(16))) __bf16   v16bf;
typedef __attribute__((ext_vector_type(8)))  float    v8f;
typedef __attribute__((ext_vector_type(8)))  unsigned v8u;

#define B_  32
#define NV_ 2048
#define NL_ 128
#define D_  512
#define KC  64     // K chunk staged in LDS (bf16)
#define RB  128    // video rows per workgroup

// Assemble a 16-element bf16 fragment from two 16-byte LDS chunks.
__device__ __forceinline__ v16bf frag16(const __bf16* p0, const __bf16* p1) {
    uint4 a = *(const uint4*)p0;
    uint4 b = *(const uint4*)p1;
    v8u r;
    r[0]=a.x; r[1]=a.y; r[2]=a.z; r[3]=a.w;
    r[4]=b.x; r[5]=b.y; r[6]=b.z; r[7]=b.w;
    return __builtin_bit_cast(v16bf, r);
}

__global__ __launch_bounds__(256) void chamfer_tile(
    const float* __restrict__ video, const float* __restrict__ lang,
    const float* __restrict__ mask_v, const float* __restrict__ mask_l,
    float* __restrict__ partRow,   // [B][16]   sum of (row mins * mask_v) per row-block
    float* __restrict__ colMin)    // [B][16][NL] partial column mins per row-block
{
    __shared__ __align__(16) __bf16 Ahi[RB*KC], Alo[RB*KC];
    __shared__ __align__(16) __bf16 Bhi[NL_*KC], Blo[NL_*KC];
    __shared__ float aN[RB], bN[NL_], mV[RB], mL[NL_];
    __shared__ float colbuf[NL_*8];
    __shared__ float rowsum[8];

    const int rb   = blockIdx.x;          // row block (0..15)
    const int b    = blockIdx.y;          // batch
    const int tid  = threadIdx.x;
    const int lane = tid & 31;
    const int wave = tid >> 5;
    const int row2 = tid >> 1;            // 0..127 (staging/norm row)
    const int hf   = tid & 1;             // which 256-element half of the row

    const size_t vbase = ((size_t)b * NV_ + (size_t)rb * RB) * D_;
    const size_t lbase = (size_t)b * NL_ * D_;

    // ---- masks to LDS ----
    if (tid < RB)  mV[tid] = mask_v[(size_t)b * NV_ + (size_t)rb * RB + tid];
    if (tid < NL_) mL[tid] = mask_l[(size_t)b * NL_ + tid];

    // ---- squared norms (f32, exact) ----
    {
        const float4* vp = (const float4*)(video + vbase + (size_t)row2 * D_ + hf * 256);
        const float4* lp = (const float4*)(lang  + lbase + (size_t)row2 * D_ + hf * 256);
        float sv = 0.f, sl = 0.f;
        #pragma unroll 8
        for (int i = 0; i < 64; ++i) {
            float4 x = vp[i]; sv += x.x*x.x + x.y*x.y + x.z*x.z + x.w*x.w;
            float4 y = lp[i]; sl += y.x*y.x + y.y*y.y + y.z*y.z + y.w*y.w;
        }
        sv += __shfl_xor(sv, 1, 32);
        sl += __shfl_xor(sl, 1, 32);
        if (hf == 0) { aN[row2] = sv; bN[row2] = sl; }
    }

    // ---- WMMA accumulators: wave owns rows [16*wave,16*wave+16) x all 128 cols ----
    v8f c[8] = {};

    const int K0a  = (lane & 16) ? 16 : 0;   // A frag K base (contiguous k = 16h + s, matches B)
    const int K0b  = (lane & 16) ? 16 : 0;   // B frag K base (contiguous k = 16h + s, same as A)
    const int mrow = wave * 16 + (lane & 15);

    for (int kc = 0; kc < D_ / KC; ++kc) {
        __syncthreads();
        // stage K-chunk, convert f32 -> bf16 hi/lo
        {
            const float4* vp = (const float4*)(video + vbase + (size_t)row2 * D_ + kc * KC + hf * 32);
            const float4* lp = (const float4*)(lang  + lbase + (size_t)row2 * D_ + kc * KC + hf * 32);
            const int o = row2 * KC + hf * 32;
            #pragma unroll
            for (int i = 0; i < 8; ++i) {
                float4 x = vp[i];
                float4 y = lp[i];
                float xs[4] = {x.x, x.y, x.z, x.w};
                float ys[4] = {y.x, y.y, y.z, y.w};
                #pragma unroll
                for (int j = 0; j < 4; ++j) {
                    __bf16 h = (__bf16)xs[j];
                    Ahi[o + 4*i + j] = h;
                    Alo[o + 4*i + j] = (__bf16)(xs[j] - (float)h);
                    __bf16 g = (__bf16)ys[j];
                    Bhi[o + 4*i + j] = g;
                    Blo[o + 4*i + j] = (__bf16)(ys[j] - (float)g);
                }
            }
        }
        __syncthreads();
        // 2 K-steps of 32 per chunk; 3 wmma per fragment (hi*hi + hi*lo + lo*hi)
        #pragma unroll
        for (int ks = 0; ks < 2; ++ks) {
            const int abase = mrow * KC + ks * 32 + K0a;
            v16bf ah = frag16(&Ahi[abase], &Ahi[abase + 8]);
            v16bf al = frag16(&Alo[abase], &Alo[abase + 8]);
            #pragma unroll
            for (int nf = 0; nf < 8; ++nf) {
                const int n = nf * 16 + (lane & 15);
                const int bbase = n * KC + ks * 32 + K0b;
                v16bf bh = frag16(&Bhi[bbase], &Bhi[bbase + 8]);
                v16bf bl = frag16(&Blo[bbase], &Blo[bbase + 8]);
                c[nf] = __builtin_amdgcn_wmma_f32_16x16x32_bf16(false, ah, false, bh, (short)0, c[nf], false, false);
                c[nf] = __builtin_amdgcn_wmma_f32_16x16x32_bf16(false, ah, false, bl, (short)0, c[nf], false, false);
                c[nf] = __builtin_amdgcn_wmma_f32_16x16x32_bf16(false, al, false, bh, (short)0, c[nf], false, false);
            }
        }
    }
    __syncthreads();

    // ---- epilogue: pairwise = a + b - 2G, mask -> +FLT_MAX, row/col mins ----
    float bcol[8], mlc[8];
    #pragma unroll
    for (int nf = 0; nf < 8; ++nf) {
        const int col = nf * 16 + (lane & 15);
        bcol[nf] = bN[col];
        mlc[nf]  = mL[col];
    }

    float colm[8];
    #pragma unroll
    for (int nf = 0; nf < 8; ++nf) colm[nf] = FLT_MAX;
    float rsum = 0.f;

    #pragma unroll
    for (int r = 0; r < 8; ++r) {
        // C/D layout: VGPR r, lanes 0-15 -> M=r, lanes 16-31 -> M=r+8; N = lane%16
        const int row = wave * 16 + r + ((lane & 16) ? 8 : 0);
        const float arow = aN[row];
        const float mvr  = mV[row];
        float rmin = FLT_MAX;
        #pragma unroll
        for (int nf = 0; nf < 8; ++nf) {
            float P = arow + bcol[nf] - 2.f * c[nf][r];
            P = (mvr > 0.f && mlc[nf] > 0.f) ? P : FLT_MAX;
            rmin = fminf(rmin, P);
            colm[nf] = fminf(colm[nf], P);
        }
        #pragma unroll
        for (int s = 1; s < 16; s <<= 1)
            rmin = fminf(rmin, __shfl_xor(rmin, s, 32));
        if ((lane & 15) == 0 && mvr > 0.f) rsum += rmin;   // lanes 0 & 16 hold half-mins
    }

    // column mins: fold the two lane halves, then across waves via LDS
    #pragma unroll
    for (int nf = 0; nf < 8; ++nf)
        colm[nf] = fminf(colm[nf], __shfl_xor(colm[nf], 16, 32));
    if (lane < 16) {
        #pragma unroll
        for (int nf = 0; nf < 8; ++nf)
            colbuf[(nf * 16 + lane) * 8 + wave] = colm[nf];
    }
    rsum += __shfl_xor(rsum, 16, 32);
    if (lane == 0) rowsum[wave] = rsum;
    __syncthreads();

    if (tid < NL_) {
        float m = colbuf[tid * 8];
        #pragma unroll
        for (int w = 1; w < 8; ++w) m = fminf(m, colbuf[tid * 8 + w]);
        colMin[((size_t)b * 16 + rb) * NL_ + tid] = m;
    }
    if (tid == 0) {
        float s = 0.f;
        for (int w = 0; w < 8; ++w) s += rowsum[w];
        partRow[b * 16 + rb] = s;
    }
}

__global__ __launch_bounds__(128) void chamfer_final(
    const float* __restrict__ mask_v, const float* __restrict__ mask_l,
    const float* __restrict__ partRow, const float* __restrict__ colMin,
    float* __restrict__ out)
{
    __shared__ float red[128];
    const int b = blockIdx.x, tid = threadIdx.x;

    float m = FLT_MAX;
    for (int blk = 0; blk < 16; ++blk)
        m = fminf(m, colMin[((size_t)b * 16 + blk) * NL_ + tid]);
    const float ml = mask_l[(size_t)b * NL_ + tid];

    float sl = (ml > 0.f) ? m : 0.f;   // mins_l * mask_l
    float nl = ml;                      // Nl partial
    float nv = 0.f;                     // Nv partial
    for (int i = 0; i < 16; ++i)
        nv += mask_v[(size_t)b * NV_ + tid + 128 * i];
    float rp = (tid < 16) ? partRow[b * 16 + tid] : 0.f;

    float acc[4] = {sl, nl, nv, rp};
    float tot[4];
    #pragma unroll
    for (int k = 0; k < 4; ++k) {
        red[tid] = acc[k];
        __syncthreads();
        for (int s = 64; s > 0; s >>= 1) {
            if (tid < s) red[tid] += red[tid + s];
            __syncthreads();
        }
        tot[k] = red[0];
        __syncthreads();
    }
    if (tid == 0)
        out[b] = tot[0] / tot[1] + tot[3] / tot[2];
}

extern "C" void kernel_launch(void* const* d_in, const int* in_sizes, int n_in,
                              void* d_out, int out_size, void* d_ws, size_t ws_size,
                              hipStream_t stream) {
    const float* video  = (const float*)d_in[0];
    const float* lang   = (const float*)d_in[1];
    const float* mask_v = (const float*)d_in[2];
    const float* mask_l = (const float*)d_in[3];
    float* out = (float*)d_out;

    float* partRow = (float*)d_ws;            // 32*16 floats
    float* colMin  = partRow + B_ * 16;       // 32*16*128 floats

    dim3 g2(NV_ / RB, B_);
    chamfer_tile<<<g2, 256, 0, stream>>>(video, lang, mask_v, mask_l, partRow, colMin);
    chamfer_final<<<B_, 128, 0, stream>>>(mask_v, mask_l, partRow, colMin, out);
}
